// VectorQuantizer_27298812133561
// MI455X (gfx1250) — matrix operand order repaired and checked by reading, not compiled, not run on hardware
//
#include <hip/hip_runtime.h>
#include <hip/hip_bf16.h>
#include <math.h>

// ---------------- problem constants ----------------
#define NROWS 65536   // 16 * 4096
#define FSZ   512
#define DSZ   128
#define NVARS 1024
#define TSZ   4096
#define GAMMA 0.25f

#define Q_OFF    1
#define PERP_OFF (1 + NROWS * DSZ)      // 8388609
#define K_OFF    (PERP_OFF + 1)         // 8388610

// LDS layout (dynamic): [0, 256KB) codebook bf16
//                       [256KB, +4KB) cnorm f32
//                       [+4KB, +32KB) z staging bf16 (8 waves * 16 * 128)
//                       [+32KB, +512B) per-wave k indices
#define SMEM_CB_BYTES   (NVARS * DSZ * 2)               // 262144
#define SMEM_CN_OFF     SMEM_CB_BYTES
#define SMEM_Z_OFF      (SMEM_CN_OFF + NVARS * 4)       // 266240
#define SMEM_K_OFF      (SMEM_Z_OFF + 8 * 16 * DSZ * 2) // 299008
#define SMEM_BYTES      (SMEM_K_OFF + 8 * 16 * 4)       // 299520

// ws layout (bytes): [0,4KB) cnorm | [4KB,8KB) counts | [8KB] sse |
//                    [16KB, +256KB) codebook bf16 | [+256KB, +128KB) W bf16
#define WS_CB_OFF  16384
#define WS_W_OFF   (WS_CB_OFF + NVARS * DSZ * 2)

typedef __attribute__((ext_vector_type(16))) __bf16        v16bf;
typedef __attribute__((ext_vector_type(8)))  float         v8f;
typedef __attribute__((ext_vector_type(4)))  float         v4f;
typedef __attribute__((ext_vector_type(4)))  unsigned int  v4u;

union Frag {
    v16bf b;
    v4u   q[2];
};

__device__ __forceinline__ v8f zero_v8f() {
    v8f z;
#pragma unroll
    for (int i = 0; i < 8; ++i) z[i] = 0.0f;
    return z;
}

// gfx1250 async copy: global -> LDS, tracked by ASYNCcnt
__device__ __forceinline__ void async_load_b128(unsigned ldsOff, const void* gptr) {
    unsigned long long ga = (unsigned long long)gptr;
    asm volatile("global_load_async_to_lds_b128 %0, %1, off"
                 :: "v"(ldsOff), "v"(ga) : "memory");
}
__device__ __forceinline__ void wait_async0() {
    asm volatile("s_wait_asynccnt 0x0" ::: "memory");
}

// ---------------- prep 1: cnorm, zero accumulators ----------------
__global__ void vq_prep(const float* __restrict__ cb,
                        float* __restrict__ cnorm,
                        float* __restrict__ counts,
                        float* __restrict__ sse) {
    int t = threadIdx.x;                      // 1024 threads, one per codeword
    const float* row = cb + (size_t)t * DSZ;
    float s = 0.f;
#pragma unroll 8
    for (int j = 0; j < DSZ; ++j) { float c = row[j]; s += c * c; }
    cnorm[t]  = s;
    counts[t] = 0.f;
    if (t == 0) *sse = 0.f;
}

// ---------------- prep 2: f32 -> bf16 conversion of codebook and W ----------
__global__ void vq_cvt(const float* __restrict__ cb, const float* __restrict__ W,
                       __bf16* __restrict__ cbbf, __bf16* __restrict__ Wbf) {
    int i = blockIdx.x * 256 + threadIdx.x;
    if (i < NVARS * DSZ) cbbf[i] = (__bf16)cb[i];
    if (i < DSZ * FSZ)   Wbf[i]  = (__bf16)W[i];
}

// ---------------- fused main kernel ----------------
__global__ void __launch_bounds__(256, 1)
vq_main(const float* __restrict__ x, const __bf16* __restrict__ Wbf,
        const float* __restrict__ bias, const float* __restrict__ cb,
        const float* __restrict__ cnorm, const __bf16* __restrict__ cbbf,
        float* __restrict__ counts, float* __restrict__ sse_acc,
        float* __restrict__ out_q, float* __restrict__ out_k) {
    extern __shared__ char smem[];
    __bf16* cblds = (__bf16*)smem;                       // 1024 x 128 bf16
    float*  cnlds = (float*)(smem + SMEM_CN_OFF);
    __bf16* zst   = (__bf16*)(smem + SMEM_Z_OFF);
    int*    kw    = (int*)(smem + SMEM_K_OFF);

    const int tid  = threadIdx.x;
    const int wave = tid >> 5;
    const int lane = tid & 31;
    const int l16  = lane & 15;
    const int lh   = lane >> 4;

    // ===== kick off async LDS fill of the bf16 codebook (overlaps phase A) ===
    {
        // generic LDS pointer: aperture lives in addr[63:32], low 32 bits are
        // the LDS byte offset (ISA 10.2), so truncation gives the ds address.
        unsigned base = (unsigned)(unsigned long long)(void*)smem;
        const char* src = (const char*)cbbf;
#pragma unroll 8
        for (int i = 0; i < 64; ++i) {
            unsigned off = (unsigned)(tid + i * 256) * 16u;
            async_load_b128(base + off, src + off);
        }
        for (int i = tid; i < NVARS; i += 256) cnlds[i] = cnorm[i];
    }

    // ================= Phase A: z = x @ W^T + b  (no LDS needed) ============
    const int    rowBase = blockIdx.x * 128 + wave * 16;
    const float* xrow    = x + (size_t)(rowBase + l16) * FSZ;

    v8f acc[8];
#pragma unroll
    for (int nt = 0; nt < 8; ++nt) acc[nt] = zero_v8f();

#pragma unroll 2
    for (int kc = 0; kc < FSZ / 32; ++kc) {
        // ---- batch all loads for this K-chunk first (one wait, many loads) --
        // A fragment source: x row m=l16, K = {lh*8..+7, 16+lh*8..+7}
        const float* ap = xrow + kc * 32 + lh * 8;
        v4f a0 = *(const v4f*)(ap);
        v4f a1 = *(const v4f*)(ap + 4);
        v4f a2 = *(const v4f*)(ap + 16);
        v4f a3 = *(const v4f*)(ap + 20);
        // B fragments: W^T chunk, B[k][n] = W[n][k]; lane holds col n=l16,
        // K = {lh*8..+7, 16+lh*8..+7} of Wbf row n (L2-resident).
        Frag Bw[8];
#pragma unroll
        for (int nt = 0; nt < 8; ++nt) {
            const __bf16* bp = Wbf + (nt * 16 + l16) * FSZ + kc * 32 + lh * 8;
            Bw[nt].q[0] = *(const v4u*)(bp);
            Bw[nt].q[1] = *(const v4u*)(bp + 16);
        }
        // ---- convert A to bf16 (v_cvt_pk_bf16_f32) while loads are in flight
        Frag A;
        A.b[0]  = (__bf16)a0.x; A.b[1]  = (__bf16)a0.y; A.b[2]  = (__bf16)a0.z; A.b[3]  = (__bf16)a0.w;
        A.b[4]  = (__bf16)a1.x; A.b[5]  = (__bf16)a1.y; A.b[6]  = (__bf16)a1.z; A.b[7]  = (__bf16)a1.w;
        A.b[8]  = (__bf16)a2.x; A.b[9]  = (__bf16)a2.y; A.b[10] = (__bf16)a2.z; A.b[11] = (__bf16)a2.w;
        A.b[12] = (__bf16)a3.x; A.b[13] = (__bf16)a3.y; A.b[14] = (__bf16)a3.z; A.b[15] = (__bf16)a3.w;
        // ---- 8 back-to-back WMMAs -----------------------------------------
#pragma unroll
        for (int nt = 0; nt < 8; ++nt)
            acc[nt] = __builtin_amdgcn_wmma_f32_16x16x32_bf16(
                false, A.b, false, Bw[nt].b, (short)0, acc[nt], false, false);
    }
    // bias: C-layout column n = nt*16 + l16, identical for all 8 VGPR rows
#pragma unroll
    for (int nt = 0; nt < 8; ++nt) {
        float bv = bias[nt * 16 + l16];
#pragma unroll
        for (int r = 0; r < 8; ++r) acc[nt][r] += bv;
    }

    // ===== stage z (bf16) to per-wave LDS region (C-layout -> row-major) ====
    __bf16* zw = zst + wave * 16 * DSZ;
#pragma unroll
    for (int nt = 0; nt < 8; ++nt) {
        int n = nt * 16 + l16;
#pragma unroll
        for (int r = 0; r < 8; ++r) {
            int m = r + 8 * lh;                 // C-layout: M = r + 8*(lane>=16)
            zw[m * DSZ + n] = (__bf16)acc[nt][r];
        }
    }

    // codebook async copy must be complete and visible to all waves
    wait_async0();
    __syncthreads();

    // ================= Phase B: argmin_v (|c|^2 - 2 z.c) ====================
    Frag ZA[4];
#pragma unroll
    for (int kc = 0; kc < 4; ++kc) {
        const __bf16* zp = zw + l16 * DSZ + kc * 32 + lh * 8;
        ZA[kc].q[0] = *(const v4u*)(zp);        // K = lh*8 .. +7
        ZA[kc].q[1] = *(const v4u*)(zp + 16);   // K = 16+lh*8 .. +7
    }

    float best[8]; int bidx[8];
#pragma unroll
    for (int r = 0; r < 8; ++r) { best[r] = 3.4e38f; bidx[r] = 0; }

    Frag B0[4], B1[4];
#pragma unroll
    for (int kc = 0; kc < 4; ++kc) {            // preload vt = 0
        const __bf16* cp = cblds + l16 * DSZ + kc * 32 + lh * 8;
        B0[kc].q[0] = *(const v4u*)(cp);
        B0[kc].q[1] = *(const v4u*)(cp + 16);
    }

    for (int vt = 0; vt < NVARS / 16; vt += 2) {
        // prefetch vt+1 while computing vt (fixed buffers, no dynamic indexing)
#pragma unroll
        for (int kc = 0; kc < 4; ++kc) {
            const __bf16* cp = cblds + ((vt + 1) * 16 + l16) * DSZ + kc * 32 + lh * 8;
            B1[kc].q[0] = *(const v4u*)(cp);
            B1[kc].q[1] = *(const v4u*)(cp + 16);
        }
        {
            v8f dot = zero_v8f();
#pragma unroll
            for (int kc = 0; kc < 4; ++kc)
                dot = __builtin_amdgcn_wmma_f32_16x16x32_bf16(
                    false, ZA[kc].b, false, B0[kc].b, (short)0, dot, false, false);
            int   v  = vt * 16 + l16;
            float cn = cnlds[v];
#pragma unroll
            for (int r = 0; r < 8; ++r) {
                float score = cn - 2.0f * dot[r];
                if (score < best[r]) { best[r] = score; bidx[r] = v; }
            }
        }
        if (vt + 2 < NVARS / 16) {              // prefetch vt+2 into B0
#pragma unroll
            for (int kc = 0; kc < 4; ++kc) {
                const __bf16* cp = cblds + ((vt + 2) * 16 + l16) * DSZ + kc * 32 + lh * 8;
                B0[kc].q[0] = *(const v4u*)(cp);
                B0[kc].q[1] = *(const v4u*)(cp + 16);
            }
        }
        {
            v8f dot = zero_v8f();
#pragma unroll
            for (int kc = 0; kc < 4; ++kc)
                dot = __builtin_amdgcn_wmma_f32_16x16x32_bf16(
                    false, ZA[kc].b, false, B1[kc].b, (short)0, dot, false, false);
            int   v  = (vt + 1) * 16 + l16;
            float cn = cnlds[v];
#pragma unroll
            for (int r = 0; r < 8; ++r) {
                float score = cn - 2.0f * dot[r];
                if (score < best[r]) { best[r] = score; bidx[r] = v; }
            }
        }
    }

    // cross-lane argmin within each 16-lane group (xor masks < 16 stay in group)
#pragma unroll
    for (int off = 1; off < 16; off <<= 1) {
#pragma unroll
        for (int r = 0; r < 8; ++r) {
            float ob = __shfl_xor(best[r], off, 32);
            int   oi = __shfl_xor(bidx[r], off, 32);
            if (ob < best[r] || (ob == best[r] && oi < bidx[r])) { best[r] = ob; bidx[r] = oi; }
        }
    }

    if (l16 == 0) {        // lane 0 handles rows 0..7, lane 16 rows 8..15
#pragma unroll
        for (int r = 0; r < 8; ++r) {
            int m = r + 8 * lh;
            int n = rowBase + m;
            kw[wave * 16 + m] = bidx[r];
            out_k[n] = (float)bidx[r];
            atomicAdd(&counts[bidx[r]], 1.0f);
        }
    }
    __syncthreads();

    // ===== output: quantized = codebook[k] (transposed), SSE for loss =======
    const int m    = l16;                       // wave-local row == t offset
    const int nrow = rowBase + m;
    const int bI   = nrow >> 12;
    const int tI   = nrow & (TSZ - 1);
    const int kv   = kw[wave * 16 + m];
    const float* crow = cb + (size_t)kv * DSZ;  // full-precision gather
    float* qp = out_q + (size_t)bI * DSZ * TSZ + tI;

    float sse = 0.f;
#pragma unroll 4
    for (int j = 0; j < 64; ++j) {
        int   d  = 2 * j + lh;
        float qv = crow[d];
        float zv = (float)zw[m * DSZ + d];
        float df = zv - qv;
        sse += df * df;
        qp[(size_t)d * TSZ] = qv;               // lanes 0..15: contiguous t's
    }
#pragma unroll
    for (int off = 1; off < 32; off <<= 1) sse += __shfl_xor(sse, off, 32);
    if (lane == 0) atomicAdd(sse_acc, sse);
}

// ---------------- finalize: perplexity + latent loss ----------------
__global__ void vq_finalize(const float* __restrict__ counts,
                            const float* __restrict__ sse,
                            float* __restrict__ out) {
    __shared__ float red[1024];
    int t = threadIdx.x;
    float p = counts[t] * (1.0f / (float)NROWS);
    red[t] = p * logf(p + 1e-7f);
    __syncthreads();
    for (int s = 512; s > 0; s >>= 1) {
        if (t < s) red[t] += red[t + s];
        __syncthreads();
    }
    if (t == 0) {
        out[PERP_OFF] = expf(-red[0]);
        float mse = (*sse) / (float)((size_t)NROWS * DSZ);
        out[0] = (GAMMA + 1.0f) * mse;   // e_loss == q_loss numerically
    }
}

// ---------------- host entry ----------------
extern "C" void kernel_launch(void* const* d_in, const int* in_sizes, int n_in,
                              void* d_out, int out_size, void* d_ws, size_t ws_size,
                              hipStream_t stream) {
    const float* x  = (const float*)d_in[0];
    const float* W  = (const float*)d_in[1];
    const float* b  = (const float*)d_in[2];
    const float* cb = (const float*)d_in[3];
    float* out = (float*)d_out;

    float*  cnorm  = (float*)d_ws;                        // 1024
    float*  counts = cnorm + NVARS;                       // 1024
    float*  sse    = (float*)((char*)d_ws + 8192);        // 1
    __bf16* cbbf   = (__bf16*)((char*)d_ws + WS_CB_OFF);  // 1024*128
    __bf16* Wbf    = (__bf16*)((char*)d_ws + WS_W_OFF);   // 128*512

    (void)in_sizes; (void)n_in; (void)out_size; (void)ws_size;

    hipFuncSetAttribute((const void*)vq_main,
                        hipFuncAttributeMaxDynamicSharedMemorySize, SMEM_BYTES);

    vq_prep<<<1, 1024, 0, stream>>>(cb, cnorm, counts, sse);
    vq_cvt<<<512, 256, 0, stream>>>(cb, W, cbbf, Wbf);
    vq_main<<<NROWS / 128, 256, SMEM_BYTES, stream>>>(
        x, Wbf, b, cb, cnorm, cbbf, counts, sse, out + Q_OFF, out + K_OFF);
    vq_finalize<<<1, 1024, 0, stream>>>(counts, sse, out);
}
